// SelfAttention_62036507623531
// MI455X (gfx1250) — matrix operand order repaired and checked by reading, not compiled, not run on hardware
//
#include <hip/hip_runtime.h>
#include <hip/hip_bf16.h>

typedef __attribute__((ext_vector_type(16))) _Float16 v16h;
typedef __attribute__((ext_vector_type(8)))  _Float16 v8h;
typedef __attribute__((ext_vector_type(4)))  _Float16 v4h;
typedef __attribute__((ext_vector_type(4)))  float    v4f;
typedef __attribute__((ext_vector_type(8)))  float    v8f;
typedef __attribute__((ext_vector_type(4)))  unsigned int u32x4;
typedef __attribute__((ext_vector_type(8)))  int      i32x8;
typedef __attribute__((ext_vector_type(4)))  int      i32x4;

#define WMMA_F16(a, b, c) \
  __builtin_amdgcn_wmma_f32_16x16x32_f16(false, (a), false, (b), (short)0, (c), false, false)

// Tensor Data Mover availability (device pass only; host pass parses fallback)
#if defined(__HIP_DEVICE_COMPILE__)
# if __has_builtin(__builtin_amdgcn_tensor_load_to_lds)
#  define USE_TDM 1
# endif
#endif
#ifndef USE_TDM
# define USE_TDM 0
#endif
#if USE_TDM && __has_include(<hip/amd_detail/amd_gfx1250_TDM.h>)
# define TDM_6ARG 1   // therock/clang-23 headers present -> 6-arg builtin
#else
# define TDM_6ARG 0   // ROCm 7.2 / clang-22 -> 5-arg builtin
#endif

__device__ __forceinline__ v16h cat16(v8h lo, v8h hi) {
  return __builtin_shufflevector(lo, hi, 0, 1, 2, 3, 4, 5, 6, 7, 8, 9, 10, 11, 12, 13, 14, 15);
}

// ---------------------------------------------------------------------------
// fp32 -> f16 conversion, 4-wide
// ---------------------------------------------------------------------------
__global__ __launch_bounds__(256) void cvt_f32_f16(const float* __restrict__ s,
                                                   _Float16* __restrict__ d, int n4) {
  int i = blockIdx.x * blockDim.x + threadIdx.x;
  if (i < n4) {
    v4f x = ((const v4f*)s)[i];
    ((v4h*)d)[i] = __builtin_convertvector(x, v4h);
  }
}

// ---------------------------------------------------------------------------
// C[M,E] = A[M,K] * W[E,K]^T + bias[E]   (Linear: x @ W.T + b)
// Block = 8 waves, 128(M) x 64(E). Weight slab staged into LDS in 4 K-stages
// of 64x256 f16 via the Tensor Data Mover (TDM) so all 8 waves share one
// fetch of W (8x L2 traffic reduction vs direct global B-fragments).
// LDS row layout: 256 halves + 48B pad -> 560B stride (140 dw, 12 mod 64
// banks) => B-fragment reads hit all banks, ~conflict-free.
// mode 0: f16 scatter to [B,H,N,Dh]   (Q, K)
// mode 2: f16 scatter to [B,H,Dh,N]   (V transposed for PV B-fragments)
// mode 3: f32 row-major [M,E]         (final projection -> d_out)
// Fixed sizes: B=4, N=1024, H=16, Dh=64.
// ---------------------------------------------------------------------------
#define LDS_ROW 280   // halves per LDS weight row: 256 data + 24 pad (48B)

__global__ __launch_bounds__(256) void gemm_nt_f16(
    const _Float16* __restrict__ A, const _Float16* __restrict__ W,
    const float* __restrict__ bias, void* __restrict__ dst,
    int M, int K, int E, int mode) {
  __shared__ _Float16 wlds[64 * LDS_ROW];  // 35,840 B

  const int tid  = threadIdx.x;
  const int lane = tid & 31;
  const int wave = tid >> 5;
  const int l15  = lane & 15;
  const int sel  = lane >> 4;  // 0 or 1

  const int m0 = blockIdx.y * 128 + wave * 16;
  const int e0 = blockIdx.x * 64;

  v8f acc[4] = {};
  // A fragment base: row = m0 + l15 ; K chunks {sel*8, 16+sel*8}
  const _Float16* Ap = A + (size_t)(m0 + l15) * K + sel * 8;

  for (int ks = 0; ks < 4; ++ks) {
    const _Float16* Wstage = W + (size_t)e0 * K + ks * 256;
#if USE_TDM
    if (wave == 0) {
      // Tensor DMA descriptor (CDNA5 ISA ch.8): 2D tile 256(K) x 64(E) f16,
      // tensor_dim0_stride = K = 1024 elems, LDS pad 12 dwords per 128 dwords.
      u32x4 g0;
      i32x8 g1;
      i32x4 gz = {0, 0, 0, 0};
      const unsigned long long ga = (unsigned long long)Wstage;
      const unsigned int ldsb = (unsigned int)(unsigned long long)&wlds[0];
      g0[0] = 1u;                                                   // count=1
      g0[1] = ldsb;                                                 // lds_addr
      g0[2] = (unsigned int)ga;                                     // global_addr lo
      g0[3] = (unsigned int)((ga >> 32) & 0x01FFFFFFu) | 0x80000000u; // ga hi | type=2
      g1[0] = 0x17910000;   // data_size=2B, pad_enable, pad_interval=128dw, pad_amount=12dw
      g1[1] = 0;            // tensor_dim0 = 1<<20 (low16 -> [31:16] = 0)
      g1[2] = 0x00000010;   // tensor_dim0 hi16 = 0x10 ; tensor_dim1 low16 = 0
      g1[3] = 0x01000010;   // tensor_dim1 hi16 = 0x10 ; tile_dim0 = 256
      g1[4] = 64;           // tile_dim1 = 64 ; tile_dim2 = 0
      g1[5] = 1024;         // tensor_dim0_stride = K
      g1[6] = 0;
      g1[7] = 0;
#if TDM_6ARG
      i32x8 z8 = {0, 0, 0, 0, 0, 0, 0, 0};
      __builtin_amdgcn_tensor_load_to_lds(g0, g1, gz, gz, z8, 0);
#else
      __builtin_amdgcn_tensor_load_to_lds(g0, g1, gz, gz, 0);
#endif
      __builtin_amdgcn_s_wait_tensorcnt(0);
    }
#else
    // Fallback: cooperative copy into the same padded layout (ds_store_b128)
    for (int i = tid; i < 64 * 32; i += 256) {
      const int row = i >> 5, c = (i & 31) * 8;
      *(v8h*)&wlds[row * LDS_ROW + c] = *(const v8h*)(Wstage + (size_t)row * K + c);
    }
#endif
    __syncthreads();

    const _Float16* Aks = Ap + ks * 256;
    for (int k0 = 0; k0 < 256; k0 += 32) {
      const int kk = k0 + sel * 8;  // B-fragment K chunk for this half-wave
      v16h a = cat16(*(const v8h*)(Aks + k0), *(const v8h*)(Aks + k0 + 16));
      const _Float16* Lb = wlds + kk;
      v16h b0 = cat16(*(const v8h*)(Lb + (0 * 16 + l15) * LDS_ROW),
                      *(const v8h*)(Lb + (0 * 16 + l15) * LDS_ROW + 16));
      v16h b1 = cat16(*(const v8h*)(Lb + (1 * 16 + l15) * LDS_ROW),
                      *(const v8h*)(Lb + (1 * 16 + l15) * LDS_ROW + 16));
      v16h b2 = cat16(*(const v8h*)(Lb + (2 * 16 + l15) * LDS_ROW),
                      *(const v8h*)(Lb + (2 * 16 + l15) * LDS_ROW + 16));
      v16h b3 = cat16(*(const v8h*)(Lb + (3 * 16 + l15) * LDS_ROW),
                      *(const v8h*)(Lb + (3 * 16 + l15) * LDS_ROW + 16));
      acc[0] = WMMA_F16(a, b0, acc[0]);
      acc[1] = WMMA_F16(a, b1, acc[1]);
      acc[2] = WMMA_F16(a, b2, acc[2]);
      acc[3] = WMMA_F16(a, b3, acc[3]);
    }
    __syncthreads();
  }

  // Epilogue. C layout: lanes 0-15 hold M=r (vgpr r), lanes 16-31 hold M=8+r; N=l15.
  const int rbase = sel * 8;
#pragma unroll
  for (int g = 0; g < 4; ++g) {
    const int e = e0 + g * 16 + l15;
    const float bv = bias[e];
#pragma unroll
    for (int r = 0; r < 8; ++r) {
      const int m = m0 + rbase + r;
      const float val = acc[g][r] + bv;
      if (mode == 3) {
        ((float*)dst)[(size_t)m * E + e] = val;
      } else {
        const int b = m >> 10, n = m & 1023;   // x rows are [B,N] flat
        const int h = e >> 6,  d = e & 63;     // cols are [H,Dh] flat
        const _Float16 hv = (_Float16)val;
        if (mode == 2)
          ((_Float16*)dst)[(((size_t)(b * 16 + h)) * 64 + d) * 1024 + n] = hv;  // [BH,Dh,N]
        else
          ((_Float16*)dst)[(((size_t)(b * 16 + h)) * 1024 + n) * 64 + d] = hv;  // [BH,N,Dh]
      }
    }
  }
}

// ---------------------------------------------------------------------------
// Attention: 1 wave per (bh, 16-row Q block). softmax(Q K^T / sqrt(N)) V.
// Qh,Kh: [64, 1024, 64] f16 ; Vt: [64, 64, 1024] f16 ; O: [64, 1024, 64] f16.
// ---------------------------------------------------------------------------
__global__ __launch_bounds__(32) void attention_kernel(
    const _Float16* __restrict__ Qh, const _Float16* __restrict__ Kh,
    const _Float16* __restrict__ Vt, _Float16* __restrict__ O) {
  __shared__ _Float16 sbuf[16 * 1024];  // 32 KB score/prob row-block

  const int lane = threadIdx.x & 31;
  const int l15  = lane & 15;
  const int sel  = lane >> 4;
  const int qb   = blockIdx.x & 63;
  const int bh   = blockIdx.x >> 6;
  const int qbase = qb * 16;

  // --- Q fragments: 2 K-steps covering Dh=64 ---
  const _Float16* qrow = Qh + ((size_t)bh * 1024 + qbase + l15) * 64;
  v16h qa0, qa1;
  {
    v8h lo0 = *(const v8h*)(qrow + 0 * 32 + sel * 16);
    v8h hi0 = *(const v8h*)(qrow + 0 * 32 + 8 + sel * 16);
    v8h lo1 = *(const v8h*)(qrow + 1 * 32 + sel * 16);
    v8h hi1 = *(const v8h*)(qrow + 1 * 32 + 8 + sel * 16);
    qa0 = cat16(lo0, hi0);
    qa1 = cat16(lo1, hi1);
  }

  // --- Scores: S[16, 1024] = Q16xDh * K^T -> LDS (f16, unscaled) ---
  for (int nt = 0; nt < 64; ++nt) {
    const _Float16* krow = Kh + ((size_t)bh * 1024 + nt * 16 + l15) * 64 + sel * 16;
    v16h b0 = *(const v16h*)(krow);
    v16h b1 = *(const v16h*)(krow + 32);
    v8f c = {};
    c = WMMA_F16(qa0, b0, c);
    c = WMMA_F16(qa1, b1, c);
    const int col = nt * 16 + l15;
#pragma unroll
    for (int r = 0; r < 8; ++r)
      sbuf[(size_t)(sel * 8 + r) * 1024 + col] = (_Float16)c[r];
  }

  // --- Softmax over 1024 cols per row; lane pair (l15, l15+16) splits a row ---
  // Reference quirk: scale = N^-0.5 = 1/32 (seq len, not head dim).
  const float scale = 0.03125f;
  _Float16* srow = sbuf + (size_t)l15 * 1024 + sel * 512;
  float mx = -3.0e38f;
  for (int j = 0; j < 512; ++j) mx = fmaxf(mx, (float)srow[j]);
  mx = fmaxf(mx, __shfl_xor(mx, 16, 32));
  float sum = 0.0f;
  for (int j = 0; j < 512; ++j) {
    float p = __expf(scale * ((float)srow[j] - mx));
    sum += p;
    srow[j] = (_Float16)p;
  }
  sum += __shfl_xor(sum, 16, 32);
  const float inv = 1.0f / sum;
  for (int j = 0; j < 512; ++j) srow[j] = (_Float16)((float)srow[j] * inv);

  // --- O[16, 64] = P[16,1024] * V[1024,64] ; A-fragments from LDS ---
  v8f oa[4] = {};
  for (int kt = 0; kt < 32; ++kt) {
    const _Float16* prow = sbuf + (size_t)l15 * 1024 + kt * 32 + sel * 16;
    v8h plo = *(const v8h*)(prow);
    v8h phi = *(const v8h*)(prow + 8);
    v16h a  = cat16(plo, phi);
    const _Float16* vb = Vt + ((size_t)bh * 64) * 1024 + kt * 32 + sel * 16;
    v16h b0 = *(const v16h*)(vb + (size_t)(0 * 16 + l15) * 1024);
    v16h b1 = *(const v16h*)(vb + (size_t)(1 * 16 + l15) * 1024);
    v16h b2 = *(const v16h*)(vb + (size_t)(2 * 16 + l15) * 1024);
    v16h b3 = *(const v16h*)(vb + (size_t)(3 * 16 + l15) * 1024);
    oa[0] = WMMA_F16(a, b0, oa[0]);
    oa[1] = WMMA_F16(a, b1, oa[1]);
    oa[2] = WMMA_F16(a, b2, oa[2]);
    oa[3] = WMMA_F16(a, b3, oa[3]);
  }

  // Store O in [BH, N, Dh] (contiguous == reference's un-permuted reshape)
  _Float16* orow = O + ((size_t)bh * 1024 + qbase) * 64;
#pragma unroll
  for (int g = 0; g < 4; ++g) {
#pragma unroll
    for (int r = 0; r < 8; ++r)
      orow[(size_t)(sel * 8 + r) * 64 + g * 16 + l15] = (_Float16)oa[g][r];
  }
}

// ---------------------------------------------------------------------------
extern "C" void kernel_launch(void* const* d_in, const int* in_sizes, int n_in,
                              void* d_out, int out_size, void* d_ws, size_t ws_size,
                              hipStream_t stream) {
  (void)in_sizes; (void)n_in; (void)out_size; (void)ws_size;
  const float* x  = (const float*)d_in[0];
  const float* Wq = (const float*)d_in[1];
  const float* bq = (const float*)d_in[2];
  const float* Wk = (const float*)d_in[3];
  const float* bk = (const float*)d_in[4];
  const float* Wv = (const float*)d_in[5];
  const float* bv = (const float*)d_in[6];
  const float* Wc = (const float*)d_in[7];
  const float* bc = (const float*)d_in[8];

  const size_t SZ_X = 4096ull * 1024;  // activations (elems)
  const size_t SZ_W = 1024ull * 1024;  // weights (elems)

  _Float16* ws  = (_Float16*)d_ws;
  _Float16* xh  = ws;
  _Float16* wqh = xh + SZ_X;
  _Float16* wkh = wqh + SZ_W;
  _Float16* wvh = wkh + SZ_W;
  _Float16* wch = wvh + SZ_W;
  _Float16* qh  = wch + SZ_W;   // [B,H,N,Dh]
  _Float16* kh  = qh + SZ_X;    // [B,H,N,Dh]
  _Float16* vth = kh + SZ_X;    // [B,H,Dh,N]
  _Float16* oh  = vth + SZ_X;   // [B,H,N,Dh] == flat [4096,1024]

  cvt_f32_f16<<<(int)((SZ_X / 4 + 255) / 256), 256, 0, stream>>>(x,  xh,  (int)(SZ_X / 4));
  cvt_f32_f16<<<(int)((SZ_W / 4 + 255) / 256), 256, 0, stream>>>(Wq, wqh, (int)(SZ_W / 4));
  cvt_f32_f16<<<(int)((SZ_W / 4 + 255) / 256), 256, 0, stream>>>(Wk, wkh, (int)(SZ_W / 4));
  cvt_f32_f16<<<(int)((SZ_W / 4 + 255) / 256), 256, 0, stream>>>(Wv, wvh, (int)(SZ_W / 4));
  cvt_f32_f16<<<(int)((SZ_W / 4 + 255) / 256), 256, 0, stream>>>(Wc, wch, (int)(SZ_W / 4));

  dim3 gg(16, 32);  // E/64 x M/128
  gemm_nt_f16<<<gg, 256, 0, stream>>>(xh, wqh, bq, qh,  4096, 1024, 1024, 0);
  gemm_nt_f16<<<gg, 256, 0, stream>>>(xh, wkh, bk, kh,  4096, 1024, 1024, 0);
  gemm_nt_f16<<<gg, 256, 0, stream>>>(xh, wvh, bv, vth, 4096, 1024, 1024, 2);

  attention_kernel<<<4096, 32, 0, stream>>>(qh, kh, vth, oh);

  gemm_nt_f16<<<gg, 256, 0, stream>>>(oh, wch, bc, d_out, 4096, 1024, 1024, 3);
}
